// CrossAttention_MP_28956669509998
// MI455X (gfx1250) — matrix operand order repaired and checked by reading, not compiled, not run on hardware
//
#include <hip/hip_runtime.h>
#include <hip/hip_bf16.h>

// ---------------------------------------------------------------------------
// CrossAttention_MP for MI455X (gfx1250, wave32, WMMA bf16->f32, async-LDS)
//
// Shapes: B=8, C=512, H=W=96.  qk_scale = 0.125.
// The torch-style reshape [B,H,W,3C] -> [B,3,C,H,W] makes each qkv slice a
// contiguous flat third of the per-batch [H,W,3C] projection output:
//   q2[b] = proj(y)[positions    0..3071 ]  (p-major, o-minor)
//   k1[b] = proj(x)[positions 3072..6143 ]
//   v1[b] = proj(x)[positions 6144..9215 ]
// and slice s viewed flat as [C=512][96][96] row-major IS the [c,i,j] tensor.
// ---------------------------------------------------------------------------

typedef __bf16 bf16;
typedef __attribute__((ext_vector_type(16))) __bf16 v16bf;
typedef __attribute__((ext_vector_type(8)))  __bf16 v8bf;
typedef __attribute__((ext_vector_type(4)))  __bf16 v4bf;
typedef __attribute__((ext_vector_type(2)))  __bf16 v2bf;
typedef __attribute__((ext_vector_type(8)))  float  v8f;
typedef __attribute__((ext_vector_type(4)))  float  v4f;

#define B_       8
#define C_       512
#define H_       96
#define W_       96
#define P_       (H_ * W_)        // 9216 positions per batch
#define CO_      (3 * C_)         // 1536 projection outputs
#define SLICE_P  3072             // positions per qkv slice
#define SLICE_EL ((size_t)SLICE_P * CO_)   // 4,718,592 elems per batch-slice
#define QKSCALE  0.125f

union AF16 { v16bf v; v8bf h[2]; };

// ---- CDNA5 async global->LDS copy (ASYNCcnt), 16B per lane ---------------
__device__ __forceinline__ void async_b128(unsigned lds_byte, const bf16* g) {
  asm volatile("global_load_async_to_lds_b128 %0, %1, off"
               :: "v"(lds_byte), "v"((unsigned long long)(size_t)g)
               : "memory");
}
__device__ __forceinline__ void wait_async0() {
  asm volatile("s_wait_asynccnt 0x0" ::: "memory");
}
__device__ __forceinline__ unsigned lds_addr(const void* p) {
  return (unsigned)(size_t)p;   // low 32 bits of flat LDS addr = LDS offset
}

// ---------------------------------------------------------------------------
// Kernel 1: fp32 W_qkv [1536x512] -> bf16
// ---------------------------------------------------------------------------
__global__ __launch_bounds__(256) void wcvt_kernel(const float* __restrict__ w,
                                                   bf16* __restrict__ wb) {
  int i = (blockIdx.x * 256 + threadIdx.x) * 4;   // 786432 total, exact
  v4f f = *(const v4f*)(w + i);
  v4bf r;
#pragma unroll
  for (int e = 0; e < 4; ++e) r[e] = (bf16)f[e];
  *(v4bf*)(wb + i) = r;
}

// ---------------------------------------------------------------------------
// Kernel 2: fused QKV projection.  out[p,o] = sum_c in[b,c,p] * W[o,c]
// Only the needed thirds are computed (y: p<3072 -> q2; x: p>=3072 -> k1/v1).
// Block = 256 thr (8 waves), tile 64(p) x 64(o), K chunks of 32,
// double-buffered LDS; W tile staged via async-LDS, A tile cvt'd in VALU.
// ---------------------------------------------------------------------------
__global__ __launch_bounds__(256)
void qkv_proj_kernel(const float* __restrict__ x, const float* __restrict__ y,
                     const bf16* __restrict__ wb,
                     bf16* __restrict__ q2, bf16* __restrict__ k1,
                     bf16* __restrict__ v1) {
  __shared__ bf16 As[2][64][32];   // A tile:  As[buf][p][c]
  __shared__ bf16 Bt[2][64][32];   // B^T tile: Bt[buf][o][c]

  const int tid  = threadIdx.x;
  const int lane = tid & 31;
  const int wv   = tid >> 5;    // 0..7
  const int wm   = wv >> 1;     // row stripe 0..3  (16 rows each)
  const int wn   = wv & 1;      // col half  0..1   (32 cols each)

  const int g0 = blockIdx.x * 64;          // global position-tile start
  const int b  = g0 / P_;
  const int pl = g0 - b * P_;              // batch-local position
  const int s  = pl / SLICE_P;             // qkv slice 0/1/2
  const int ps = pl - s * SLICE_P;         // slice-local position
  const float* src  = (s == 0) ? y : x;
  bf16*        dst  = (s == 0) ? q2 : (s == 1) ? k1 : v1;
  const float* srcb = src + (size_t)b * C_ * P_ + pl;
  bf16*        dstb = dst + (size_t)b * SLICE_EL + (size_t)ps * CO_;

  const int o0 = blockIdx.y * 64;

  v8f acc0 = {}; v8f acc1 = {};

  // staging work split
  const int a_cp = (tid & 15) * 2;     // channel-pair base 0..30
  const int a_pb = (tid >> 4) * 4;     // position chunk    0..60
  const int b_n  = tid >> 2;           // output row 0..63
  const int b_k  = (tid & 3) * 8;      // channel 0/8/16/24

  const int ar  = wm * 16 + (lane & 15);
  const int akb = (lane < 16) ? 0 : 8;     // ISA 16-bit A layout K split
  const int bkb = (lane < 16) ? 0 : 8;     // ISA 16-bit B layout K split

  auto stage = [&](int buf, int c0) {
    // A: transpose channel-major fp32 -> [p][c] bf16, packed b32 stores
    const float* gp = srcb + (size_t)(c0 + a_cp) * P_ + a_pb;
    v4f fa = *(const v4f*)(gp);
    v4f fb = *(const v4f*)(gp + P_);
#pragma unroll
    for (int e = 0; e < 4; ++e) {
      v2bf pr; pr[0] = (bf16)fa[e]; pr[1] = (bf16)fb[e];
      *(v2bf*)&As[buf][a_pb + e][a_cp] = pr;
    }
    // B^T: pure bf16 copy, async global->LDS (bypasses VGPRs)
    async_b128(lds_addr(&Bt[buf][b_n][b_k]),
               wb + (size_t)(o0 + b_n) * C_ + c0 + b_k);
  };

  stage(0, 0);
  wait_async0();
  __syncthreads();

  for (int kc = 0; kc < 16; ++kc) {
    const int cur = kc & 1;
    if (kc + 1 < 16) stage(cur ^ 1, (kc + 1) * 32);

    // A fragment (lane=row; elems 0-7 = K kb..kb+7, 8-15 = K kb+16..kb+23)
    AF16 af;
    af.h[0] = *(const v8bf*)&As[cur][ar][akb];
    af.h[1] = *(const v8bf*)&As[cur][ar][akb + 16];

#pragma unroll
    for (int t = 0; t < 2; ++t) {
      const int bn = wn * 32 + t * 16 + (lane & 15);
      AF16 bfg;
      bfg.h[0] = *(const v8bf*)&Bt[cur][bn][bkb];
      bfg.h[1] = *(const v8bf*)&Bt[cur][bn][bkb + 16];
      if (t == 0)
        acc0 = __builtin_amdgcn_wmma_f32_16x16x32_bf16(false, af.v, false,
                                                       bfg.v, (short)0, acc0,
                                                       false, false);
      else
        acc1 = __builtin_amdgcn_wmma_f32_16x16x32_bf16(false, af.v, false,
                                                       bfg.v, (short)0, acc1,
                                                       false, false);
    }
    wait_async0();     // next-buffer async writes landed (own wave)
    __syncthreads();   // all waves done reading cur / writing cur^1
  }

  // store C (f32 acc -> bf16); lane=col, vgpr r -> row r (+8 hi half)
  const int m_half = (lane < 16) ? 0 : 8;
  const int ncol   = lane & 15;
#pragma unroll
  for (int r = 0; r < 8; ++r) {
    const int m = wm * 16 + r + m_half;
    bf16* orow = dstb + (size_t)m * CO_ + o0 + wn * 32;
    orow[ncol]      = (bf16)acc0[r];
    orow[16 + ncol] = (bf16)acc1[r];
  }
}

// ---------------------------------------------------------------------------
// Kernel 3: per-(b,c) attention.  96x96 S = Q K^T * 0.125, softmax, O = P V.
// Block = 192 thr (6 waves); wave w owns row stripe [16w,16w+16).
// Q/K staged via async-LDS copies; V transposed with packed b32 stores.
// LDS: Qs (aliased as P after softmax), Ks, Vt = 54 KB.
// ---------------------------------------------------------------------------
__global__ __launch_bounds__(192)
void attn_kernel(const bf16* __restrict__ q2, const bf16* __restrict__ k1,
                 const bf16* __restrict__ v1, float* __restrict__ out) {
  __shared__ bf16 Qs[96][96];   // Q, then reused for P (same wave-stripe rows)
  __shared__ bf16 Ks[96][96];   // K as-is: row t, col j  (== B^T layout)
  __shared__ bf16 Vt[96][96];   // V transposed: Vt[w][j] (== B^T layout)

  const int tid  = threadIdx.x;
  const int lane = tid & 31;
  const int wv   = tid >> 5;                 // 0..5

  const int bc = blockIdx.x;                 // b*C + c
  const int b  = bc / C_;
  const int c  = bc - b * C_;
  const size_t base = (size_t)b * SLICE_EL + (size_t)c * (H_ * W_);
  const bf16* qg = q2 + base;
  const bf16* kg = k1 + base;
  const bf16* vg = v1 + base;

  // ---- Q/K: async global->LDS, 1152 x 16B chunks each
  const unsigned qb = lds_addr(&Qs[0][0]);
  const unsigned kb = lds_addr(&Ks[0][0]);
  for (int idx = tid; idx < 1152; idx += 192) {
    async_b128(qb + idx * 16, qg + idx * 8);
    async_b128(kb + idx * 16, kg + idx * 8);
  }
  // ---- V: transpose in LDS, row pairs -> packed b32 column stores
  for (int idx = tid; idx < 576; idx += 192) {
    const int j  = (idx / 12) * 2;     // even source row
    const int w8 = (idx % 12) * 8;     // source col start
    v8bf r0 = *(const v8bf*)(vg + (size_t)j * W_ + w8);
    v8bf r1 = *(const v8bf*)(vg + (size_t)(j + 1) * W_ + w8);
#pragma unroll
    for (int e = 0; e < 8; ++e) {
      v2bf pr; pr[0] = r0[e]; pr[1] = r1[e];
      *(v2bf*)&Vt[w8 + e][j] = pr;
    }
  }
  wait_async0();
  __syncthreads();

  const int i0     = wv * 16;
  const int arow   = i0 + (lane & 15);
  const int akb    = (lane < 16) ? 0 : 8;
  const int bkb    = (lane < 16) ? 0 : 8;
  const int m_half = (lane < 16) ? 0 : 8;
  const int ncol   = lane & 15;

  // ---- S = Q @ K^T  (6 tiles across, K=96 in 3 chunks of 32)
  v8f S[6];
#pragma unroll
  for (int t = 0; t < 6; ++t) S[t] = (v8f){};
#pragma unroll
  for (int j0 = 0; j0 < 96; j0 += 32) {
    AF16 af;
    af.h[0] = *(const v8bf*)&Qs[arow][j0 + akb];
    af.h[1] = *(const v8bf*)&Qs[arow][j0 + akb + 16];
#pragma unroll
    for (int t = 0; t < 6; ++t) {
      const int n = t * 16 + (lane & 15);
      AF16 bfg;
      bfg.h[0] = *(const v8bf*)&Ks[n][j0 + bkb];
      bfg.h[1] = *(const v8bf*)&Ks[n][j0 + bkb + 16];
      S[t] = __builtin_amdgcn_wmma_f32_16x16x32_bf16(false, af.v, false, bfg.v,
                                                     (short)0, S[t], false,
                                                     false);
    }
  }

  // ---- row softmax in registers: row m lives in one 16-lane half-wave
#pragma unroll
  for (int r = 0; r < 8; ++r) {
    float m = -3.0e38f;
#pragma unroll
    for (int t = 0; t < 6; ++t) m = fmaxf(m, S[t][r]);
    m *= QKSCALE;  // scale > 0: commutes with max
#pragma unroll
    for (int off = 1; off < 16; off <<= 1) m = fmaxf(m, __shfl_xor(m, off, 32));
    float sum = 0.0f;
#pragma unroll
    for (int t = 0; t < 6; ++t) {
      const float p = __expf(S[t][r] * QKSCALE - m);
      S[t][r] = p;
      sum += p;
    }
#pragma unroll
    for (int off = 1; off < 16; off <<= 1) sum += __shfl_xor(sum, off, 32);
    const float inv  = 1.0f / sum;
    const int   irow = i0 + r + m_half;     // rows of own stripe only
#pragma unroll
    for (int t = 0; t < 6; ++t)
      Qs[irow][t * 16 + ncol] = (bf16)(S[t][r] * inv);
  }
  __syncthreads();

  // ---- O = P @ V  (A from Qs-as-P rows of own stripe, B from Vt)
  v8f O[6];
#pragma unroll
  for (int t = 0; t < 6; ++t) O[t] = (v8f){};
#pragma unroll
  for (int j0 = 0; j0 < 96; j0 += 32) {
    AF16 af;
    af.h[0] = *(const v8bf*)&Qs[arow][j0 + akb];
    af.h[1] = *(const v8bf*)&Qs[arow][j0 + akb + 16];
#pragma unroll
    for (int t = 0; t < 6; ++t) {
      const int n = t * 16 + (lane & 15);
      AF16 bfg;
      bfg.h[0] = *(const v8bf*)&Vt[n][j0 + bkb];
      bfg.h[1] = *(const v8bf*)&Vt[n][j0 + bkb + 16];
      O[t] = __builtin_amdgcn_wmma_f32_16x16x32_bf16(false, af.v, false, bfg.v,
                                                     (short)0, O[t], false,
                                                     false);
    }
  }

  // ---- write fp32 output [B,C,H,W]
  float* ob = out + (size_t)bc * (H_ * W_);
#pragma unroll
  for (int r = 0; r < 8; ++r) {
    const int irow = i0 + r + m_half;
    float* orow = ob + (size_t)irow * W_;
#pragma unroll
    for (int t = 0; t < 6; ++t) orow[t * 16 + ncol] = O[t][r];
  }
}

// ---------------------------------------------------------------------------
// Launcher.  Workspace layout (bytes), total ~228 MB:
//   [0, 1.5MB)   W_qkv in bf16
//   then q2, k1, v1: each B * 3072 * 1536 bf16 = 72 MB
// ---------------------------------------------------------------------------
extern "C" void kernel_launch(void* const* d_in, const int* in_sizes, int n_in,
                              void* d_out, int out_size, void* d_ws,
                              size_t ws_size, hipStream_t stream) {
  (void)in_sizes; (void)n_in; (void)out_size; (void)ws_size;
  const float* x  = (const float*)d_in[0];
  const float* y  = (const float*)d_in[1];
  const float* wq = (const float*)d_in[2];
  float* out = (float*)d_out;

  char* ws = (char*)d_ws;
  bf16* wb = (bf16*)ws;
  size_t off = (((size_t)CO_ * C_ * sizeof(bf16)) + 255) & ~(size_t)255;
  bf16* q2 = (bf16*)(ws + off); off += (size_t)B_ * SLICE_EL * sizeof(bf16);
  bf16* k1 = (bf16*)(ws + off); off += (size_t)B_ * SLICE_EL * sizeof(bf16);
  bf16* v1 = (bf16*)(ws + off);

  wcvt_kernel<<<(CO_ * C_) / 1024, 256, 0, stream>>>(wq, wb);

  dim3 gproj((B_ * P_) / 64, CO_ / 64);   // 1152 x 24
  qkv_proj_kernel<<<gproj, 256, 0, stream>>>(x, y, wb, q2, k1, v1);

  attn_kernel<<<B_ * C_, 192, 0, stream>>>(q2, k1, v1, out);
}
